// Model_31086973288633
// MI455X (gfx1250) — compile-verified, matrix operand order checked by reading, not run on hardware
//
#include <hip/hip_runtime.h>
#include <math.h>

// ---------------------------------------------------------------------------
// Autoformer-style forecaster for gfx1250 (MI455X).
// All GEMMs use v_wmma_f32_16x16x32_f16 (f16 in, f32 accum).  Weights are
// pre-transposed to N x K once so the GEMM B-operand is always row-per-N,
// giving fully vectorized LDS staging (b128 loads + b128 stores, no guards
// in the K loop).  The FFT autocorrelation is replaced by the equivalent
// Gram-matrix form:
//   mean_{h,e} corr[b,tau] = (1/D) * sum_t <q[b,t,:], k[b,(t-tau)%L,:]>
// i.e. a batched Q*K^T GEMM followed by wrapped-diagonal sums.
// ---------------------------------------------------------------------------

#define BATCH 32
#define SEQL  336
#define LABL  168
#define PREDL 96
#define DECL  264      // LABEL_LEN + PRED_LEN
#define NC    7
#define NMARK 4
#define DM    512
#define DFF   2048
#define RH    64
#define TOPK_ENC 17    // int(3*ln(336))
#define TOPK_DEC 16    // int(3*ln(264))

typedef _Float16 h8  __attribute__((ext_vector_type(8)));
typedef _Float16 h16 __attribute__((ext_vector_type(16)));
typedef float    f8  __attribute__((ext_vector_type(8)));
typedef float    f4v __attribute__((ext_vector_type(4)));

// ---------------------------------------------------------------------------
// WMMA GEMM:  C[m,n] = act( sum_k A[m,k]*Bnk[n,k] + bias[n] ) + Cadd[m,n]
//   A row-major MxK (lda);  Bnk row-major NxK (ldb)  [i.e. B already "T"]
//   batched via blockIdx.z with 64-bit strides.
// Block: 256 threads = 8 waves.  Tile 64(M) x 64(N), K-step 32.
// Wave w: M-tile (w>>1)*16, N-strip (w&1)*32 -> two 16x16 WMMAs per K-step
// sharing one A fragment.
//
// Edge tiles: staging rows are CLAMPED rather than zero-filled.  Out-of-range
// A rows only influence output rows >= M and out-of-range B rows only
// influence output columns >= N, both of which are masked at the epilogue
// store; K is always a multiple of TK.  This keeps the K-loop free of any
// exec-mask manipulation (WMMA requires EXEC all-ones anyway).
// ---------------------------------------------------------------------------
#define TM 64
#define TN 64
#define TK 32

__global__ __launch_bounds__(256) void gemm_f16_wmma(
    const float* __restrict__ A, const float* __restrict__ Bnk,
    const float* __restrict__ bias, const float* __restrict__ Cadd,
    float* __restrict__ C,
    int M, int N, int K,
    long long sA, long long sB, long long sC,
    int lda, int ldb, int ldc, int act)
{
  __shared__ _Float16 As[TM][TK + 8];
  __shared__ _Float16 Bs[TN][TK + 8];

  const int tid  = threadIdx.x;
  const int lane = tid & 31;
  const int wave = tid >> 5;
  const int wm   = (wave >> 1) * 16;  // 0,16,32,48
  const int wn   = (wave & 1) * 32;   // 0,32
  const int bm   = blockIdx.y * TM;
  const int bn   = blockIdx.x * TN;

  const float* Ab = A   + (long long)blockIdx.z * sA;
  const float* Bb = Bnk + (long long)blockIdx.z * sB;
  float*       Cb = C   + (long long)blockIdx.z * sC;
  const float* Rb = Cadd ? (Cadd + (long long)blockIdx.z * sC) : (const float*)0;

  // staging assignment: 8 contiguous K-floats per thread, clamped rows
  const int srow = tid >> 2;          // 0..63
  const int scol = (tid & 3) * 8;     // 0,8,16,24
  const int arow = (bm + srow) < M ? (bm + srow) : (M - 1);
  const int brow = (bn + srow) < N ? (bn + srow) : (N - 1);
  const float* Ap = Ab + (long long)arow * lda + scol;
  const float* Bp = Bb + (long long)brow * ldb + scol;

  const int g = lane >> 4;   // K-half selector
  const int r = lane & 15;

  f8 acc0 = {}, acc1 = {};

  for (int k0 = 0; k0 < K; k0 += TK) {
    // ---- stage A and B tiles (f32 -> f16, one b128 LDS store each)
    const f4v* pa = (const f4v*)(Ap + k0);
    const f4v* pb = (const f4v*)(Bp + k0);
    f4v a0 = pa[0], a1 = pa[1];
    f4v b0 = pb[0], b1 = pb[1];
    h8 ha, hb;
    #pragma unroll
    for (int j = 0; j < 4; ++j) {
      ha[j] = (_Float16)a0[j]; ha[4 + j] = (_Float16)a1[j];
      hb[j] = (_Float16)b0[j]; hb[4 + j] = (_Float16)b1[j];
    }
    *(h8*)&As[srow][scol] = ha;
    *(h8*)&Bs[srow][scol] = hb;
    __syncthreads();

    // ---- fragments per ISA wave32 layout; two WMMAs share the A fragment
    {
      h8 fa0 = *(const h8*)&As[wm + r][g * 8];
      h8 fa1 = *(const h8*)&As[wm + r][16 + g * 8];
      h8 fb0 = *(const h8*)&Bs[wn + r][g * 8];
      h8 fb1 = *(const h8*)&Bs[wn + r][16 + g * 8];
      h8 fc0 = *(const h8*)&Bs[wn + 16 + r][g * 8];
      h8 fc1 = *(const h8*)&Bs[wn + 16 + r][16 + g * 8];
      h16 av, bv0, bv1;
      #pragma unroll
      for (int i = 0; i < 8; ++i) {
        av[i]  = fa0[i]; av[8 + i]  = fa1[i];
        bv0[i] = fb0[i]; bv0[8 + i] = fb1[i];
        bv1[i] = fc0[i]; bv1[8 + i] = fc1[i];
      }
      acc0 = __builtin_amdgcn_wmma_f32_16x16x32_f16(
          false, av, false, bv0, (short)0, acc0, false, false);
      acc1 = __builtin_amdgcn_wmma_f32_16x16x32_f16(
          false, av, false, bv1, (short)0, acc1, false, false);
    }
    __syncthreads();
  }

  // ---- epilogue: bias + activation + residual, bounds-guarded store
  const int rbase = bm + wm + g * 8;
  #pragma unroll
  for (int t = 0; t < 2; ++t) {
    int col = bn + wn + t * 16 + r;
    if (col < N) {
      float bvv = bias ? bias[col] : 0.f;
      f8 acc = t ? acc1 : acc0;
      #pragma unroll
      for (int r2 = 0; r2 < 8; ++r2) {
        int row = rbase + r2;
        if (row < M) {
          float v = acc[r2] + bvv;
          if (act == 1) v = 0.5f * v * (1.f + erff(v * 0.70710678118f)); // exact GELU
          long long idx = (long long)row * ldc + col;
          if (Rb) v += Rb[idx];
          Cb[idx] = v;
        }
      }
    }
  }
}

// ---------------------------------------------------------------------------
// Elementwise / reduction kernels
// ---------------------------------------------------------------------------
__global__ void k_transpose(const float* __restrict__ in, float* __restrict__ out,
                            int K, int N) {  // in KxN -> out NxK
  long long i = (long long)blockIdx.x * 256 + threadIdx.x;
  if (i >= (long long)K * N) return;
  int k = (int)(i / N), n = (int)(i % N);
  out[(long long)n * K + k] = in[i];
}

__global__ void k_fill_zero(float* p, long long n) {
  long long i = (long long)blockIdx.x * 256 + threadIdx.x;
  if (i < n) p[i] = 0.f;
}

__global__ void k_acc_add(float* dst, const float* src, long long n) {
  long long i = (long long)blockIdx.x * 256 + threadIdx.x;
  if (i < n) dst[i] += src[i];
}

// mean over time axis: x (B,L,C) -> out (B,C)
__global__ void k_mean_time(const float* x, float* out, int B_, int L, int Cc) {
  int i = blockIdx.x * 256 + threadIdx.x;
  if (i >= B_ * Cc) return;
  int b = i / Cc, c = i % Cc;
  float s = 0.f;
  for (int l = 0; l < L; ++l) s += x[((long long)b * L + l) * Cc + c];
  out[i] = s / (float)L;
}

// series_decomp: moving average k=25 with replicate padding
__global__ void k_decomp(const float* x, float* seas, float* trend,
                         int B_, int L, int Cc) {
  long long n = (long long)B_ * L * Cc;
  long long i = (long long)blockIdx.x * 256 + threadIdx.x;
  if (i >= n) return;
  int c = (int)(i % Cc);
  long long bl = i / Cc;
  int l = (int)(bl % L);
  int b = (int)(bl / L);
  float s = 0.f;
  for (int j = -12; j <= 12; ++j) {
    int ll = l + j;
    ll = ll < 0 ? 0 : (ll >= L ? L - 1 : ll);
    s += x[((long long)b * L + ll) * Cc + c];
  }
  float t = s * (1.f / 25.f);
  if (trend) trend[i] = t;
  if (seas)  seas[i]  = x[i] - t;
}

// build decoder seasonal_init / trend_init (B,264,7)
__global__ void k_init_dec(const float* se, const float* te, const float* meanc,
                           float* seas_init, float* trend_init) {
  int i = blockIdx.x * 256 + threadIdx.x;
  if (i >= BATCH * DECL * NC) return;
  int c = i % NC;
  int l = (i / NC) % DECL;
  int b = i / (DECL * NC);
  if (l < LABL) {
    long long src = ((long long)b * SEQL + (SEQL - LABL + l)) * NC + c;
    seas_init[i]  = se[src];
    trend_init[i] = te[src];
  } else {
    seas_init[i]  = 0.f;
    trend_init[i] = meanc[b * NC + c];
  }
}

// embedding: circ_conv3(x, tok_w) + mark @ time_w -> out (B,L,512)
__global__ void k_embed(const float* x, const float* mark,
                        const float* tokw, const float* timew,
                        float* out, int B_, int L, int Cin) {
  long long n = (long long)B_ * L * DM;
  long long i = (long long)blockIdx.x * 256 + threadIdx.x;
  if (i >= n) return;
  int d = (int)(i & (DM - 1));
  long long bl = i >> 9;
  int l = (int)(bl % L);
  int b = (int)(bl / L);
  float s = 0.f;
  for (int t = 0; t < 3; ++t) {
    int ll = l + t - 1; if (ll < 0) ll += L; if (ll >= L) ll -= L;
    const float* xr = x + ((long long)b * L + ll) * Cin;
    const float* w  = tokw + (long long)t * Cin * DM + d;
    for (int c = 0; c < Cin; ++c) s += xr[c] * w[c * DM];
  }
  const float* mr = mark + ((long long)b * L + l) * NMARK;
  for (int m = 0; m < NMARK; ++m) s += mr[m] * timew[m * DM + d];
  out[i] = s;
}

// mean_value[b,tau] = (1/512) * sum_t scores[b,t,(t-tau)%L]
__global__ void k_mean_value(const float* sc, float* mv, int B_, int L) {
  int i = blockIdx.x * 256 + threadIdx.x;
  if (i >= B_ * L) return;
  int tau = i % L, b = i / L;
  const float* s = sc + (long long)b * L * L;
  float acc = 0.f;
  for (int t = 0; t < L; ++t) {
    int j = t - tau; if (j < 0) j += L;
    acc += s[(long long)t * L + j];
  }
  mv[i] = acc * (1.f / 512.f);
}

// top-k over batch-mean of mean_value (single block)
__global__ void k_topk(const float* mv, int* idx, int B_, int L, int k) {
  __shared__ float mb[SEQL + 16];
  int t = threadIdx.x;
  if (t < L) {
    float s = 0.f;
    for (int b = 0; b < B_; ++b) s += mv[b * L + t];
    mb[t] = s;
  }
  __syncthreads();
  if (t == 0) {
    for (int i = 0; i < k; ++i) {
      int best = 0; float bv = -1e30f;
      for (int tau = 0; tau < L; ++tau)
        if (mb[tau] > bv) { bv = mb[tau]; best = tau; }
      idx[i] = best;
      mb[best] = -1e30f;
    }
  }
}

// per-batch softmax over the k selected lags
__global__ void k_weights(const float* mv, const int* idx, float* w,
                          int B_, int L, int k) {
  int b = blockIdx.x * blockDim.x + threadIdx.x;
  if (b >= B_) return;
  float r[24];
  float mx = -1e30f;
  for (int i = 0; i < k; ++i) { r[i] = mv[b * L + idx[i]]; mx = fmaxf(mx, r[i]); }
  float s = 0.f;
  for (int i = 0; i < k; ++i) { r[i] = expf(r[i] - mx); s += r[i]; }
  float inv = 1.f / s;
  for (int i = 0; i < k; ++i) w[b * 32 + i] = r[i] * inv;
}

// agg[b,t,d] = sum_i v[b,(t+idx[i])%L,d] * w[b,i]
__global__ void k_agg(const float* v, const int* idx, const float* w,
                      float* out, int B_, int L, int k) {
  long long n = (long long)B_ * L * DM;
  long long i = (long long)blockIdx.x * 256 + threadIdx.x;
  if (i >= n) return;
  int d = (int)(i & (DM - 1));
  long long bl = i >> 9;
  int t = (int)(bl % L);
  int b = (int)(bl / L);
  const float* vb = v + (long long)b * L * DM;
  float s = 0.f;
  for (int j = 0; j < k; ++j) {
    int tt = t + idx[j]; if (tt >= L) tt -= L;
    s += vb[(long long)tt * DM + d] * w[b * 32 + j];
  }
  out[i] = s;
}

// per-row layernorm over channels (block per (b,l) row)
__global__ __launch_bounds__(256) void k_ln_rows(const float* x, const float* wl,
                                                 const float* bl, float* out) {
  __shared__ float red[256];
  int row = blockIdx.x, t = threadIdx.x;
  const float* xr = x + (long long)row * DM;
  float s = 0.f;
  for (int d = t; d < DM; d += 256) s += xr[d];
  red[t] = s; __syncthreads();
  for (int o = 128; o > 0; o >>= 1) { if (t < o) red[t] += red[t + o]; __syncthreads(); }
  float mu = red[0] / (float)DM;
  __syncthreads();
  float s2 = 0.f;
  for (int d = t; d < DM; d += 256) { float dd = xr[d] - mu; s2 += dd * dd; }
  red[t] = s2; __syncthreads();
  for (int o = 128; o > 0; o >>= 1) { if (t < o) red[t] += red[t + o]; __syncthreads(); }
  float inv = rsqrtf(red[0] / (float)DM + 1e-5f);
  for (int d = t; d < DM; d += 256)
    out[(long long)row * DM + d] = (xr[d] - mu) * inv * wl[d] + bl[d];
}

// per-(b,d) mean over time
__global__ void k_colmean(const float* x, float* cm, int B_, int L) {
  int i = blockIdx.x * 256 + threadIdx.x;
  if (i >= B_ * DM) return;
  int b = i / DM, d = i % DM;
  float s = 0.f;
  for (int l = 0; l < L; ++l) s += x[((long long)b * L + l) * DM + d];
  cm[i] = s / (float)L;
}

__global__ void k_submean(const float* x, const float* cm, float* out,
                          int B_, int L) {
  long long n = (long long)B_ * L * DM;
  long long i = (long long)blockIdx.x * 256 + threadIdx.x;
  if (i >= n) return;
  int d = (int)(i & (DM - 1));
  int b = (int)((i >> 9) / L);
  out[i] = x[i] - cm[b * DM + d];
}

// trend circ_conv3 (512 -> 7), accumulates into trend7
__global__ void k_trendconv(const float* tin, const float* w, float* trend7,
                            int B_, int L) {
  int i = blockIdx.x * 256 + threadIdx.x;
  if (i >= B_ * L * NC) return;
  int c = i % NC;
  int l = (i / NC) % L;
  int b = i / (NC * L);
  float s = 0.f;
  for (int t = 0; t < 3; ++t) {
    int ll = l + t - 1; if (ll < 0) ll += L; if (ll >= L) ll -= L;
    const float* xr = tin + ((long long)b * L + ll) * DM;
    const float* wr = w + (long long)t * DM * NC + c;
    for (int d = 0; d < DM; ++d) s += xr[d] * wr[d * NC];
  }
  trend7[i] += s;
}

// final projection + trend add, emitting only the last PRED_LEN steps
__global__ void k_proj_out(const float* xln, const float* trend7,
                           const float* pw, const float* pb, float* out) {
  int i = blockIdx.x * 256 + threadIdx.x;
  if (i >= BATCH * PREDL * NC) return;
  int c = i % NC;
  int l = (i / NC) % PREDL;
  int b = i / (PREDL * NC);
  int ld = LABL + l;
  const float* xr = xln + ((long long)b * DECL + ld) * DM;
  float s = pb[c];
  for (int d = 0; d < DM; ++d) s += xr[d] * pw[d * NC + c];
  out[i] = s + trend7[((long long)b * DECL + ld) * NC + c];
}

// LSTM head: input is hn repeated SEQL times, hidden 64; fc(64->1) per step
__global__ __launch_bounds__(256) void k_lstm(
    const float* enc, const float* wih, const float* whh,
    const float* bih, const float* bhh,
    const float* fcw, const float* fcb, float* out) {
  __shared__ float h[RH], c[RH], g[4 * RH], hn[DM];
  int b = blockIdx.x, t = threadIdx.x;
  for (int d = t; d < DM; d += 256)
    hn[d] = enc[((long long)b * SEQL + (SEQL - 1)) * DM + d];
  if (t < RH) { h[t] = 0.f; c[t] = 0.f; }
  __syncthreads();
  // input projection is constant across steps
  float xw;
  {
    float s = bih[t] + bhh[t];
    const float* wr = wih + (long long)t * DM;
    for (int d = 0; d < DM; ++d) s += wr[d] * hn[d];
    xw = s;
  }
  for (int step = 0; step < SEQL; ++step) {
    float s = xw;
    const float* wr = whh + (long long)t * RH;
    for (int j = 0; j < RH; ++j) s += wr[j] * h[j];
    g[t] = s;
    __syncthreads();
    if (t < RH) {
      float ig = 1.f / (1.f + expf(-g[t]));
      float fg = 1.f / (1.f + expf(-g[RH + t]));
      float gg = tanhf(g[2 * RH + t]);
      float og = 1.f / (1.f + expf(-g[3 * RH + t]));
      float cn = fg * c[t] + ig * gg;
      c[t] = cn;
      h[t] = og * tanhf(cn);
    }
    __syncthreads();
    if (t == 0) {
      float s2 = fcb[0];
      for (int j = 0; j < RH; ++j) s2 += h[j] * fcw[j];
      out[b * SEQL + step] = s2;
    }
    __syncthreads();
  }
}

// ---------------------------------------------------------------------------
// Host side
// ---------------------------------------------------------------------------
static inline int cdiv(long long a, int b) { return (int)((a + b - 1) / b); }

static void run_gemm(hipStream_t s, const float* A, const float* Bnk,
                     const float* bias, const float* Cadd, float* C,
                     int M, int N, int K,
                     long long sA, long long sB, long long sC,
                     int lda, int ldb, int ldc, int act, int batch) {
  dim3 g(cdiv(N, TN), cdiv(M, TM), batch), blk(256);
  hipLaunchKernelGGL(gemm_f16_wmma, g, blk, 0, s,
                     A, Bnk, bias, Cadd, C, M, N, K, sA, sB, sC,
                     lda, ldb, ldc, act);
}

// Q*K^T -> mean_value -> topk -> softmax -> shifted-weighted aggregation
static void autocorr_agg(hipStream_t s, const float* q, const float* k,
                         const float* v, float* sc, float* mv, int* idx,
                         float* wbuf, float* aggout, int L, int topk) {
  // K matrix is already "N x K" shaped (L x DM rows) for the Gram GEMM
  run_gemm(s, q, k, nullptr, nullptr, sc, L, L, DM,
           (long long)L * DM, (long long)L * DM, (long long)L * L,
           DM, DM, L, 0, BATCH);
  hipLaunchKernelGGL(k_mean_value, dim3(cdiv(BATCH * L, 256)), dim3(256), 0, s,
                     sc, mv, BATCH, L);
  hipLaunchKernelGGL(k_topk, dim3(1), dim3(352), 0, s, mv, idx, BATCH, L, topk);
  hipLaunchKernelGGL(k_weights, dim3(1), dim3(32), 0, s, mv, idx, wbuf, BATCH, L, topk);
  hipLaunchKernelGGL(k_agg, dim3(cdiv((long long)BATCH * L * DM, 256)), dim3(256),
                     0, s, v, idx, wbuf, aggout, BATCH, L, topk);
}

extern "C" void kernel_launch(void* const* d_in, const int* in_sizes, int n_in,
                              void* d_out, int out_size, void* d_ws, size_t ws_size,
                              hipStream_t stream) {
  (void)in_sizes; (void)n_in; (void)out_size; (void)ws_size;

  const float* x_enc      = (const float*)d_in[0];
  const float* x_mark_enc = (const float*)d_in[1];
  // d_in[2] = x_dec: only its zero-padded shape is used by the reference
  const float* x_mark_dec = (const float*)d_in[3];

  // params flattened in dict-insertion order (see reference setup_inputs)
  const float* enc_tokw  = (const float*)d_in[4];
  const float* enc_timew = (const float*)d_in[5];
  const float* dec_tokw  = (const float*)d_in[6];
  const float* dec_timew = (const float*)d_in[7];
  // enc layer l: base 8 + 10*l -> wq,bq,wk,bk,wv,bv,wo,bo,w1,w2
  const float* enc_ln_w = (const float*)d_in[28];
  const float* enc_ln_b = (const float*)d_in[29];
  // dec layer: self_attn 30..37, cross_attn 38..45, w1 46, w2 47, trend_w 48
  const float* dec_trend_w = (const float*)d_in[48];
  const float* dec_ln_w = (const float*)d_in[49];
  const float* dec_ln_b = (const float*)d_in[50];
  const float* proj_w   = (const float*)d_in[51];
  const float* proj_b   = (const float*)d_in[52];
  const float* r_wih = (const float*)d_in[53];
  const float* r_whh = (const float*)d_in[54];
  const float* r_bih = (const float*)d_in[55];
  const float* r_bhh = (const float*)d_in[56];
  const float* r_fcw = (const float*)d_in[57];
  const float* r_fcb = (const float*)d_in[58];

  // ---- workspace layout (floats)
  const long long SZ_ED = (long long)BATCH * SEQL * DM;   // 5.50M
  const long long SZ_DD = (long long)BATCH * DECL * DM;   // 4.33M
  const long long SZ_FF = (long long)BATCH * SEQL * DFF;  // 22.0M
  const long long SZ_SC = (long long)BATCH * SEQL * SEQL; // 3.61M
  float* ws = (float*)d_ws;
  long long off = 0;
  auto alloc = [&](long long n) { float* p = ws + off; off += n; return p; };
  float* enc_x = alloc(SZ_ED);
  float* tmp_a = alloc(SZ_ED);
  float* tmp_b = alloc(SZ_ED);
  float* qb    = alloc(SZ_ED);
  float* kb    = alloc(SZ_ED);
  float* vb    = alloc(SZ_ED);
  float* ffn   = alloc(SZ_FF);
  float* sc    = alloc(SZ_SC);
  float* dec_x = alloc(SZ_DD);
  float* dtr   = alloc(SZ_DD);
  float* se7   = alloc((long long)BATCH * SEQL * NC);
  float* te7   = alloc((long long)BATCH * SEQL * NC);
  float* meanc = alloc(BATCH * NC);
  float* seas_init = alloc((long long)BATCH * DECL * NC);
  float* trend7    = alloc((long long)BATCH * DECL * NC);
  float* mv    = alloc(BATCH * SEQL);
  float* wbuf  = alloc(BATCH * 32);
  float* cm    = alloc(BATCH * DM);
  int*   idxb  = (int*)alloc(32);

  // ---- pre-transpose every weight matrix to N x K form (tiny one-off cost)
  auto prepT = [&](int idx, int K, int N) -> float* {
    float* o = alloc((long long)K * N);
    hipLaunchKernelGGL(k_transpose, dim3(cdiv((long long)K * N, 256)), dim3(256),
                       0, stream, (const float*)d_in[idx], o, K, N);
    return o;
  };
  // encoder layers
  float* encT[2][6];
  for (int l = 0; l < 2; ++l) {
    int base = 8 + l * 10;
    encT[l][0] = prepT(base + 0, DM, DM);   // wq
    encT[l][1] = prepT(base + 2, DM, DM);   // wk
    encT[l][2] = prepT(base + 4, DM, DM);   // wv
    encT[l][3] = prepT(base + 6, DM, DM);   // wo
    encT[l][4] = prepT(base + 8, DM, DFF);  // w1 -> (DFF x DM)
    encT[l][5] = prepT(base + 9, DFF, DM);  // w2 -> (DM x DFF)
  }
  // decoder
  float* dsT[4];  // self wq,wk,wv,wo
  float* dcT[4];  // cross wq,wk,wv,wo
  for (int j = 0; j < 4; ++j) dsT[j] = prepT(30 + 2 * j, DM, DM);
  for (int j = 0; j < 4; ++j) dcT[j] = prepT(38 + 2 * j, DM, DM);
  float* dw1T = prepT(46, DM, DFF);
  float* dw2T = prepT(47, DFF, DM);

  const int M_enc = BATCH * SEQL;
  const int M_dec = BATCH * DECL;

  // ================= Phase 0: decomposition init =================
  hipLaunchKernelGGL(k_mean_time, dim3(cdiv(BATCH * NC, 256)), dim3(256), 0, stream,
                     x_enc, meanc, BATCH, SEQL, NC);
  hipLaunchKernelGGL(k_decomp, dim3(cdiv((long long)BATCH * SEQL * NC, 256)), dim3(256),
                     0, stream, x_enc, se7, te7, BATCH, SEQL, NC);
  hipLaunchKernelGGL(k_init_dec, dim3(cdiv(BATCH * DECL * NC, 256)), dim3(256), 0, stream,
                     se7, te7, meanc, seas_init, trend7);

  // ================= Phase 1: encoder =================
  hipLaunchKernelGGL(k_embed, dim3(cdiv(SZ_ED, 256)), dim3(256), 0, stream,
                     x_enc, x_mark_enc, enc_tokw, enc_timew, enc_x, BATCH, SEQL, NC);

  for (int l = 0; l < 2; ++l) {
    int base = 8 + l * 10;
    const float* bq = (const float*)d_in[base + 1];
    const float* bk = (const float*)d_in[base + 3];
    const float* bv = (const float*)d_in[base + 5];
    const float* bo = (const float*)d_in[base + 7];

    run_gemm(stream, enc_x, encT[l][0], bq, nullptr, qb, M_enc, DM, DM, 0,0,0, DM, DM, DM, 0, 1);
    run_gemm(stream, enc_x, encT[l][1], bk, nullptr, kb, M_enc, DM, DM, 0,0,0, DM, DM, DM, 0, 1);
    run_gemm(stream, enc_x, encT[l][2], bv, nullptr, vb, M_enc, DM, DM, 0,0,0, DM, DM, DM, 0, 1);
    autocorr_agg(stream, qb, kb, vb, sc, mv, idxb, wbuf, tmp_a, SEQL, TOPK_ENC);
    // O-projection fused with residual add
    run_gemm(stream, tmp_a, encT[l][3], bo, enc_x, tmp_b, M_enc, DM, DM, 0,0,0, DM, DM, DM, 0, 1);
    hipLaunchKernelGGL(k_decomp, dim3(cdiv(SZ_ED, 256)), dim3(256), 0, stream,
                       tmp_b, enc_x, (float*)0, BATCH, SEQL, DM);
    // FFN: gelu fused into first GEMM, residual into second
    run_gemm(stream, enc_x, encT[l][4], nullptr, nullptr, ffn, M_enc, DFF, DM, 0,0,0, DM, DM, DFF, 1, 1);
    run_gemm(stream, ffn, encT[l][5], nullptr, enc_x, tmp_a, M_enc, DM, DFF, 0,0,0, DFF, DFF, DM, 0, 1);
    hipLaunchKernelGGL(k_decomp, dim3(cdiv(SZ_ED, 256)), dim3(256), 0, stream,
                       tmp_a, enc_x, (float*)0, BATCH, SEQL, DM);
  }
  // encoder my_layernorm
  hipLaunchKernelGGL(k_ln_rows, dim3(M_enc), dim3(256), 0, stream,
                     enc_x, enc_ln_w, enc_ln_b, tmp_a);
  hipLaunchKernelGGL(k_colmean, dim3(cdiv(BATCH * DM, 256)), dim3(256), 0, stream,
                     tmp_a, cm, BATCH, SEQL);
  hipLaunchKernelGGL(k_submean, dim3(cdiv(SZ_ED, 256)), dim3(256), 0, stream,
                     tmp_a, cm, enc_x, BATCH, SEQL);

  // ================= Phase 2: decoder =================
  hipLaunchKernelGGL(k_embed, dim3(cdiv(SZ_DD, 256)), dim3(256), 0, stream,
                     seas_init, x_mark_dec, dec_tokw, dec_timew, dec_x, BATCH, DECL, NC);
  hipLaunchKernelGGL(k_fill_zero, dim3(cdiv(SZ_DD, 256)), dim3(256), 0, stream, dtr, SZ_DD);

  {
    // ---- self attention (params 30..37)
    const float* bqv = (const float*)d_in[31];
    const float* bkv = (const float*)d_in[33];
    const float* bvv = (const float*)d_in[35];
    const float* bov = (const float*)d_in[37];
    run_gemm(stream, dec_x, dsT[0], bqv, nullptr, qb, M_dec, DM, DM, 0,0,0, DM, DM, DM, 0, 1);
    run_gemm(stream, dec_x, dsT[1], bkv, nullptr, kb, M_dec, DM, DM, 0,0,0, DM, DM, DM, 0, 1);
    run_gemm(stream, dec_x, dsT[2], bvv, nullptr, vb, M_dec, DM, DM, 0,0,0, DM, DM, DM, 0, 1);
    autocorr_agg(stream, qb, kb, vb, sc, mv, idxb, wbuf, tmp_a, DECL, TOPK_DEC);
    run_gemm(stream, tmp_a, dsT[3], bov, dec_x, tmp_b, M_dec, DM, DM, 0,0,0, DM, DM, DM, 0, 1);
    hipLaunchKernelGGL(k_decomp, dim3(cdiv(SZ_DD, 256)), dim3(256), 0, stream,
                       tmp_b, dec_x, tmp_a, BATCH, DECL, DM);
    hipLaunchKernelGGL(k_acc_add, dim3(cdiv(SZ_DD, 256)), dim3(256), 0, stream,
                       dtr, tmp_a, SZ_DD);
  }
  {
    // ---- cross attention (params 38..45): K/V from first DECL steps of enc_out
    const float* bqv = (const float*)d_in[39];
    const float* bkv = (const float*)d_in[41];
    const float* bvv = (const float*)d_in[43];
    const float* bov = (const float*)d_in[45];
    run_gemm(stream, dec_x, dcT[0], bqv, nullptr, qb, M_dec, DM, DM, 0,0,0, DM, DM, DM, 0, 1);
    // batched: per-batch A = enc_out rows [0,264), stride SEQL*DM
    run_gemm(stream, enc_x, dcT[1], bkv, nullptr, kb, DECL, DM, DM,
             (long long)SEQL * DM, 0, (long long)DECL * DM, DM, DM, DM, 0, BATCH);
    run_gemm(stream, enc_x, dcT[2], bvv, nullptr, vb, DECL, DM, DM,
             (long long)SEQL * DM, 0, (long long)DECL * DM, DM, DM, DM, 0, BATCH);
    autocorr_agg(stream, qb, kb, vb, sc, mv, idxb, wbuf, tmp_a, DECL, TOPK_DEC);
    run_gemm(stream, tmp_a, dcT[3], bov, dec_x, tmp_b, M_dec, DM, DM, 0,0,0, DM, DM, DM, 0, 1);
    hipLaunchKernelGGL(k_decomp, dim3(cdiv(SZ_DD, 256)), dim3(256), 0, stream,
                       tmp_b, dec_x, tmp_a, BATCH, DECL, DM);
    hipLaunchKernelGGL(k_acc_add, dim3(cdiv(SZ_DD, 256)), dim3(256), 0, stream,
                       dtr, tmp_a, SZ_DD);
  }
  {
    // ---- decoder FFN (params 46,47)
    run_gemm(stream, dec_x, dw1T, nullptr, nullptr, ffn, M_dec, DFF, DM, 0,0,0, DM, DM, DFF, 1, 1);
    run_gemm(stream, ffn, dw2T, nullptr, dec_x, tmp_b, M_dec, DM, DFF, 0,0,0, DFF, DFF, DM, 0, 1);
    hipLaunchKernelGGL(k_decomp, dim3(cdiv(SZ_DD, 256)), dim3(256), 0, stream,
                       tmp_b, dec_x, tmp_a, BATCH, DECL, DM);
    hipLaunchKernelGGL(k_acc_add, dim3(cdiv(SZ_DD, 256)), dim3(256), 0, stream,
                       dtr, tmp_a, SZ_DD);
  }
  // trend = trend_init + circ_conv3(t1+t2+t3, trend_w)
  hipLaunchKernelGGL(k_trendconv, dim3(cdiv(BATCH * DECL * NC, 256)), dim3(256), 0, stream,
                     dtr, dec_trend_w, trend7, BATCH, DECL);
  // decoder my_layernorm
  hipLaunchKernelGGL(k_ln_rows, dim3(M_dec), dim3(256), 0, stream,
                     dec_x, dec_ln_w, dec_ln_b, tmp_a);
  hipLaunchKernelGGL(k_colmean, dim3(cdiv(BATCH * DM, 256)), dim3(256), 0, stream,
                     tmp_a, cm, BATCH, DECL);
  hipLaunchKernelGGL(k_submean, dim3(cdiv(SZ_DD, 256)), dim3(256), 0, stream,
                     tmp_a, cm, dec_x, BATCH, DECL);
  // output 1: dec_out[:, -PRED_LEN:]
  hipLaunchKernelGGL(k_proj_out, dim3(cdiv(BATCH * PREDL * NC, 256)), dim3(256), 0, stream,
                     dec_x, trend7, proj_w, proj_b, (float*)d_out);

  // ================= Phase 3: LSTM reconstruction head =================
  hipLaunchKernelGGL(k_lstm, dim3(BATCH), dim3(256), 0, stream,
                     enc_x, r_wih, r_whh, r_bih, r_bhh, r_fcw, r_fcb,
                     (float*)d_out + (long long)BATCH * PREDL * NC);
}
